// HyenaFilter_2808908611881
// MI455X (gfx1250) — matrix operand order repaired and checked by reading, not compiled, not run on hardware
//
#include <hip/hip_runtime.h>
#include <hip/hip_bf16.h>
#include <math.h>

// Problem constants (match reference)
#define B_SZ   8
#define DM     1024
#define LSEQ   8192
#define ORD    64
#define EMB    33
#define FFT_N  16384
#define FFT_LOG2 14
#define KF_STRIDE 8208      // 8193 rfft bins padded to multiple of 16
#define XSTAGE_OFF 131072u  // byte offset of x staging region in conv LDS

typedef __attribute__((ext_vector_type(16))) _Float16 v16h;
typedef __attribute__((ext_vector_type(8)))  float    v8f;

#define WMMA(A, Bf, C) \
  __builtin_amdgcn_wmma_f32_16x16x32_f16(false, (A), false, (Bf), (short)0, (C), false, false)

// ---------------------------------------------------------------------------
// WMMA fragment builders (CDNA5 16-bit layouts, CDNA5 ISA §7.12.2)
// A: 16x32 f16. lane m=lane&15, half=lane>>4. VGPR v: K = (v<4? 2v : 16+2(v-4)) + 8*half
// ---------------------------------------------------------------------------
__device__ __forceinline__ v16h make_a_frag(const float* tile /*16x64 LDS*/, int k0, int lane) {
  const int m  = lane & 15;
  const int hh = lane >> 4;
  v16h a;
#pragma unroll
  for (int v = 0; v < 8; ++v) {
    int kb = k0 + ((v & 4) ? 16 : 0) + (v & 3) * 2 + hh * 8;
    a[2 * v]     = (_Float16)tile[m * 64 + kb];
    a[2 * v + 1] = (_Float16)tile[m * 64 + kb + 1];
  }
  return a;
}

// B: 32x16 f16, row-major weights W[k][n] with row stride ldn; zero-pad k>=kmax.
// same K layout as A: VGPR v: K = (v<4? 2v : 16+2(v-4)) + 8*half, 2 K per VGPR.
__device__ __forceinline__ v16h make_b_frag(const float* __restrict__ W, int ldn,
                                            int k0, int n0, int kmax, int lane) {
  const int n  = n0 + (lane & 15);
  const int hh = lane >> 4;
  v16h b;
#pragma unroll
  for (int v = 0; v < 8; ++v) {
    int k = k0 + ((v & 4) ? 16 : 0) + (v & 3) * 2 + hh * 8;
    b[2 * v]     = (_Float16)((k     < kmax) ? W[(size_t)k       * ldn + n] : 0.f);
    b[2 * v + 1] = (_Float16)((k + 1 < kmax) ? W[(size_t)(k + 1) * ldn + n] : 0.f);
  }
  return b;
}

// One MLP hidden layer: dst[16x64] = sin(freq * (src[16x64] @ W[64x64] + bia))
__device__ __forceinline__ void mlp_layer64(const float* src, float* dst,
                                            const float* __restrict__ W,
                                            const float* __restrict__ bia,
                                            const float* __restrict__ freq, int lane) {
  const int hh = lane >> 4;
  const int nl = lane & 15;
  v16h a0 = make_a_frag(src, 0, lane);
  v16h a1 = make_a_frag(src, 32, lane);
#pragma unroll
  for (int n0 = 0; n0 < ORD; n0 += 16) {
    v8f c = {};
    c = WMMA(a0, make_b_frag(W, ORD, 0,  n0, ORD, lane), c);
    c = WMMA(a1, make_b_frag(W, ORD, 32, n0, ORD, lane), c);
    int n = n0 + nl;
    float bv = bia[n], fv = freq[n];
#pragma unroll
    for (int r = 0; r < 8; ++r)
      dst[(r + 8 * hh) * 64 + n] = __sinf(fv * (c[r] + bv));   // C layout: m=r+8*half, n=lane&15
  }
}

// ---------------------------------------------------------------------------
// Kernel 1: implicit filter MLP (WMMA) + exponential modulation -> k[d][l]
// One wave per 16 sequence rows; 4 waves per block.
// ---------------------------------------------------------------------------
__global__ __launch_bounds__(128) void hyena_mlp_kernel(
    const float* __restrict__ z, const float* __restrict__ t,
    const float* __restrict__ deltas, const float* __restrict__ freq,
    const float* __restrict__ w1, const float* __restrict__ bia1,
    const float* __restrict__ w2, const float* __restrict__ bia2,
    const float* __restrict__ w3, const float* __restrict__ bia3,
    const float* __restrict__ wout, float* __restrict__ kbuf) {
  __shared__ float smem[4][2][16 * 64];
  const int wave = threadIdx.x >> 5;
  const int lane = threadIdx.x & 31;
  const int l0   = (blockIdx.x * 4 + wave) * 16;
  float* bufA = smem[wave][0];
  float* bufB = smem[wave][1];
  const int hh = lane >> 4;
  const int nl = lane & 15;

  // stage z tile [16 x 33] into LDS, zero-padded to K=64
  for (int idx = lane; idx < 16 * 64; idx += 32) {
    int m = idx >> 6, e = idx & 63;
    bufA[idx] = (e < EMB) ? z[(size_t)(l0 + m) * EMB + e] : 0.f;
  }

  // layer 1: [16x33] @ w1[33x64], K padded to 64 (two k-steps)
  {
    v16h a0 = make_a_frag(bufA, 0, lane);
    v16h a1 = make_a_frag(bufA, 32, lane);
#pragma unroll
    for (int n0 = 0; n0 < ORD; n0 += 16) {
      v8f c = {};
      c = WMMA(a0, make_b_frag(w1, ORD, 0,  n0, EMB, lane), c);
      c = WMMA(a1, make_b_frag(w1, ORD, 32, n0, EMB, lane), c);
      int n = n0 + nl;
      float bv = bia1[n], fv = freq[n];
#pragma unroll
      for (int r = 0; r < 8; ++r)
        bufB[(r + 8 * hh) * 64 + n] = __sinf(fv * (c[r] + bv));
    }
  }
  mlp_layer64(bufB, bufA, w2, bia2, freq, lane);
  mlp_layer64(bufA, bufB, w3, bia3, freq, lane);

  // layer 4: [16x64] @ wout[64x1024], then decay modulation, scatter to k[d][l]
  {
    v16h a0 = make_a_frag(bufB, 0, lane);
    v16h a1 = make_a_frag(bufB, 32, lane);
    float tl[8];
#pragma unroll
    for (int r = 0; r < 8; ++r) tl[r] = t[l0 + r + 8 * hh];
    for (int n0 = 0; n0 < DM; n0 += 16) {
      v8f c = {};
      c = WMMA(a0, make_b_frag(wout, DM, 0,  n0, ORD, lane), c);
      c = WMMA(a1, make_b_frag(wout, DM, 32, n0, ORD, lane), c);
      int dcol = n0 + nl;
      float ad = fabsf(deltas[dcol]);
#pragma unroll
      for (int r = 0; r < 8; ++r) {
        float dec = __expf(-tl[r] * ad + 1e-16f);
        kbuf[(size_t)dcol * LSEQ + (l0 + r + 8 * hh)] = c[r] * dec;
      }
    }
  }
}

// ---------------------------------------------------------------------------
// LDS-resident 16384-point radix-2 FFT (128 KB LDS). Twiddles from a global
// table (L2-resident). inverse => conjugated twiddles, unscaled (matches
// irfft norm='forward'; the 1/N is folded into k_f).
// ---------------------------------------------------------------------------
__global__ void twiddle_kernel(float2* __restrict__ tw) {
  int i = blockIdx.x * blockDim.x + threadIdx.x;
  if (i < FFT_N / 2) {
    float ang = -2.0f * 3.14159265358979323846f * (float)i / (float)FFT_N;
    float sv, cv;
    __sincosf(ang, &sv, &cv);
    tw[i] = make_float2(cv, sv);
  }
}

__device__ __forceinline__ void fft_lds(float2* s, const float2* __restrict__ tw,
                                        int tid, int nth, bool inverse) {
  // bit-reversal permutation
  for (int i = tid; i < FFT_N; i += nth) {
    unsigned r = __brev((unsigned)i) >> (32 - FFT_LOG2);
    if (r > (unsigned)i) { float2 a = s[i]; s[i] = s[r]; s[r] = a; }
  }
  __syncthreads();
  for (int lh = 0; lh < FFT_LOG2; ++lh) {
    const int half_ = 1 << lh;
    for (int bfly = tid; bfly < (FFT_N >> 1); bfly += nth) {
      int j  = bfly & (half_ - 1);
      int i0 = ((bfly >> lh) << (lh + 1)) + j;
      int i1 = i0 + half_;
      float2 w  = tw[j << (FFT_LOG2 - 1 - lh)];
      float  wy = inverse ? -w.y : w.y;
      float2 u = s[i0];
      float2 v = s[i1];
      float tr = v.x * w.x - v.y * wy;
      float ti = v.x * wy + v.y * w.x;
      s[i0] = make_float2(u.x + tr, u.y + ti);
      s[i1] = make_float2(u.x - tr, u.y - ti);
    }
    __syncthreads();
  }
}

// Kernel 2: k_f[d] = rfft(k[d], 16384) / 16384   (half spectrum, stride KF_STRIDE)
__global__ __launch_bounds__(256) void kfft_kernel(const float* __restrict__ kbuf,
                                                   const float2* __restrict__ tw,
                                                   float2* __restrict__ kf) {
  extern __shared__ float2 s[];
  const int tid = threadIdx.x, nth = blockDim.x;
  const int d = blockIdx.x;
  const float* krow = kbuf + (size_t)d * LSEQ;
  for (int i = tid; i < FFT_N; i += nth)
    s[i] = make_float2((i < LSEQ) ? krow[i] : 0.f, 0.f);
  __syncthreads();
  fft_lds(s, tw, tid, nth, false);
  const float sc = 1.0f / (float)FFT_N;
  float2* out = kf + (size_t)d * KF_STRIDE;
  for (int i = tid; i <= LSEQ; i += nth)
    out[i] = make_float2(s[i].x * sc, s[i].y * sc);
}

// ---------------------------------------------------------------------------
// Kernel 3: per (b,d): y = ifft_unscaled(fft(x) * k_f)[:L].real + x*bias[d]
// LDS: [0,128KB) complex FFT buffer, [128KB,160KB) raw x row staged via
// GLOBAL_LOAD_ASYNC_TO_LDS_B128 (ASYNCcnt). 160KB/block -> 2 blocks per WGP.
// ---------------------------------------------------------------------------
__global__ __launch_bounds__(256) void conv_kernel(const float* __restrict__ x,
                                                   const float* __restrict__ bias,
                                                   const float2* __restrict__ tw,
                                                   const float2* __restrict__ kf,
                                                   float* __restrict__ out) {
  extern __shared__ float2 s[];
  const int tid = threadIdx.x, nth = blockDim.x;
  const int bd = blockIdx.x;
  const int d  = bd & (DM - 1);
  const float*  xrow = x  + (size_t)bd * LSEQ;
  const float2* kfr  = kf + (size_t)d  * KF_STRIDE;

  // Async DMA the 32KB x row into the LDS staging region (GV mode, B128/lane).
  // Dynamic LDS starts at offset 0 in this kernel, so the stage offset is
  // the compile-time constant XSTAGE_OFF.
  for (int c = tid; c < LSEQ / 4; c += nth) {
    unsigned lds_off = XSTAGE_OFF + (unsigned)c * 16u;
    unsigned long long ga = (unsigned long long)(uintptr_t)(xrow + c * 4);
    asm volatile("global_load_async_to_lds_b128 %0, %1, off"
                 :: "v"(lds_off), "v"(ga) : "memory");
  }

  // overlap: warm the L2->WGP path for the filter spectrum while DMA runs
  for (int i = tid * 16; i <= LSEQ; i += nth * 16)
    __builtin_prefetch((const void*)(kfr + i), 0, 1);

  asm volatile("s_wait_asynccnt 0x0" ::: "memory");
  __syncthreads();

  const float* xs = (const float*)((const char*)s + XSTAGE_OFF);
  for (int i = tid; i < FFT_N; i += nth)
    s[i] = make_float2((i < LSEQ) ? xs[i] : 0.f, 0.f);
  __syncthreads();
  fft_lds(s, tw, tid, nth, false);
  // pointwise multiply; reconstruct upper half of k_f by conjugate symmetry
  for (int i = tid; i < FFT_N; i += nth) {
    float2 K;
    if (i <= LSEQ) K = kfr[i];
    else { float2 cc = kfr[FFT_N - i]; K = make_float2(cc.x, -cc.y); }
    float2 u = s[i];
    s[i] = make_float2(u.x * K.x - u.y * K.y, u.x * K.y + u.y * K.x);
  }
  __syncthreads();
  fft_lds(s, tw, tid, nth, true);   // unscaled inverse
  const float g = bias[d];
  float* orow = out + (size_t)bd * LSEQ;
  for (int i = tid; i < LSEQ; i += nth)
    orow[i] = s[i].x + xs[i] * g;   // skip connection from LDS stage (no 2nd HBM read)
}

// ---------------------------------------------------------------------------
extern "C" void kernel_launch(void* const* d_in, const int* in_sizes, int n_in,
                              void* d_out, int out_size, void* d_ws, size_t ws_size,
                              hipStream_t stream) {
  (void)in_sizes; (void)n_in; (void)out_size; (void)ws_size;
  const float* x      = (const float*)d_in[0];
  const float* bias   = (const float*)d_in[1];
  const float* z      = (const float*)d_in[2];
  const float* t      = (const float*)d_in[3];
  const float* deltas = (const float*)d_in[4];
  const float* freq   = (const float*)d_in[5];
  const float* w1     = (const float*)d_in[6];
  const float* b1     = (const float*)d_in[7];
  const float* w2     = (const float*)d_in[8];
  const float* b2     = (const float*)d_in[9];
  const float* w3     = (const float*)d_in[10];
  const float* b3     = (const float*)d_in[11];
  const float* wout   = (const float*)d_in[12];

  // workspace layout: twiddles (64KB) | k[1024][8192] f32 (32MB) | k_f (67MB)
  char*   ws   = (char*)d_ws;
  float2* tw   = (float2*)ws;
  float*  kbuf = (float*)(ws + 65536);
  float2* kfb  = (float2*)(ws + 65536 + (size_t)DM * LSEQ * sizeof(float));

  twiddle_kernel<<<(FFT_N / 2 + 255) / 256, 256, 0, stream>>>(tw);
  hyena_mlp_kernel<<<LSEQ / 64, 128, 0, stream>>>(z, t, deltas, freq,
                                                  w1, b1, w2, b2, w3, b3, wout, kbuf);
  kfft_kernel<<<DM, 256, FFT_N * sizeof(float2), stream>>>(kbuf, tw, kfb);
  conv_kernel<<<B_SZ * DM, 256, FFT_N * sizeof(float2) + LSEQ * sizeof(float), stream>>>(
      x, bias, tw, kfb, (float*)d_out);
}
